// AGNNFeatureExtractor_14035953123699
// MI455X (gfx1250) — matrix operand order repaired and checked by reading, not compiled, not run on hardware
//
#include <hip/hip_runtime.h>

// ---------------- model constants ----------------
#define B_      128
#define NAGV_   30
#define NF_     40
#define EMB_    128
#define NNODE_  484
#define NEDGE_  1848
#define HEADS_  4
#define HD_     32
#define MEMN_   30
#define EPS_    1e-5f
#define NEG_    0.2f
#define HCH_    16384        // FFN hidden row chunk

typedef __attribute__((ext_vector_type(16))) _Float16 v16h;
typedef __attribute__((ext_vector_type(8)))  _Float16 v8h;
typedef __attribute__((ext_vector_type(4)))  _Float16 v4h;
typedef __attribute__((ext_vector_type(8)))  float    v8f;

__device__ __forceinline__ float leaky_(float x){ return x > 0.f ? x : NEG_*x; }

__device__ __forceinline__ void atomicMaxF(float* addr, float val){
  int* ai = (int*)addr;
  int old = __float_as_int(*addr);
  while (__int_as_float(old) < val){
    int assumed = old;
    old = atomicCAS(ai, assumed, __float_as_int(val));
    if (old == assumed) break;
  }
}

// ---------------- WMMA GEMM: C = act(A(MxK) * W(KxN) + bias) ----------------
// Requires N % 64 == 0 and K % 4 == 0 (true for every GEMM in this model).
// Block 256 thr = 8 waves; block tile 64x64, wave tile 16x32 (2 WMMAs/K-step).
// Staging is branchless: clamped addresses + value masking, vector loads,
// ds_store_b64 for A, f16 transpose stores for B.
__global__ __launch_bounds__(256) void k_gemm(
    const float* __restrict__ A, const float* __restrict__ W,
    const float* __restrict__ bias, float* __restrict__ C,
    int M, int N, int K, int act)
{
  __shared__ __align__(16) _Float16 As[64][32];   // [m][k]
  __shared__ __align__(16) _Float16 Bt[64][32];   // [n][k] (transposed)
  const int tid  = threadIdx.x;
  const int lane = tid & 31;
  const int w    = tid >> 5;
  const int wm   = w & 3;          // 0..3 -> 16-row slice
  const int wn   = w >> 2;         // 0..1 -> 32-col slice
  const int m0 = blockIdx.y * 64;
  const int n0 = blockIdx.x * 64;

  v8f acc0 = {}, acc1 = {};
  for (int k0 = 0; k0 < K; k0 += 32) {
    // ---- stage A tile 64x32 : 512 float4 quads, 2 per thread ----
    #pragma unroll
    for (int g = 0; g < 2; ++g) {
      int q  = tid + g*256;
      int r  = q >> 3;             // 0..63
      int c  = (q & 7) * 4;        // 0..28
      int gm = m0 + r, gk = k0 + c;
      bool ok = (gm < M) && (gk < K);
      const float4* src = (const float4*)(A + (size_t)(ok ? gm : 0) * K + (ok ? gk : 0));
      float4 v = *src;                       // always in-bounds address
      float s = ok ? 1.f : 0.f;
      v4h hv;
      hv[0] = (_Float16)(v.x*s); hv[1] = (_Float16)(v.y*s);
      hv[2] = (_Float16)(v.z*s); hv[3] = (_Float16)(v.w*s);
      *(v4h*)&As[r][c] = hv;                 // single ds_store_b64
    }
    // ---- stage B tile 64(n)x32(k), transposed: coalesced row-major loads ----
    #pragma unroll
    for (int g = 0; g < 2; ++g) {
      int q  = tid + g*256;
      int kk = q >> 4;             // 0..31
      int n  = (q & 15) * 4;       // 0..60
      int gk = k0 + kk;
      bool ok = (gk < K);
      const float4* src = (const float4*)(W + (size_t)(ok ? gk : 0) * N + n0 + n);
      float4 v = *src;
      float s = ok ? 1.f : 0.f;
      Bt[n+0][kk] = (_Float16)(v.x*s);
      Bt[n+1][kk] = (_Float16)(v.y*s);
      Bt[n+2][kk] = (_Float16)(v.z*s);
      Bt[n+3][kk] = (_Float16)(v.w*s);
    }
    __syncthreads();
    // A fragment, this kernel's contiguous K order (same as B):
    // lane<16: K {0..15}; lane>=16: K {16..31}
    const int arow = wm * 16 + (lane & 15);
    const int hi   = lane >> 4;
    const v8h a0 = *(const v8h*)&As[arow][hi * 16];
    const v8h a1 = *(const v8h*)&As[arow][hi * 16 + 8];
    v16h a;
    #pragma unroll
    for (int i = 0; i < 8; ++i) { a[i] = a0[i]; a[8+i] = a1[i]; }
    // two B fragments (cols wn*32 .. wn*32+31), K contiguous per lane half
    const int brow0 = wn * 32 + (lane & 15);
    const int brow1 = brow0 + 16;
    const v8h b00 = *(const v8h*)&Bt[brow0][hi * 16];
    const v8h b01 = *(const v8h*)&Bt[brow0][hi * 16 + 8];
    const v8h b10 = *(const v8h*)&Bt[brow1][hi * 16];
    const v8h b11 = *(const v8h*)&Bt[brow1][hi * 16 + 8];
    v16h b0v, b1v;
    #pragma unroll
    for (int i = 0; i < 8; ++i) { b0v[i]=b00[i]; b0v[8+i]=b01[i]; b1v[i]=b10[i]; b1v[8+i]=b11[i]; }
    acc0 = __builtin_amdgcn_wmma_f32_16x16x32_f16(false, a, false, b0v, (short)0, acc0, false, false);
    acc1 = __builtin_amdgcn_wmma_f32_16x16x32_f16(false, a, false, b1v, (short)0, acc1, false, false);
    __syncthreads();
  }
  // epilogue: N is exact, only M needs a guard
  const int ncol0 = n0 + wn * 32 + (lane & 15);
  const int ncol1 = ncol0 + 16;
  const float bv0 = bias ? bias[ncol0] : 0.0f;
  const float bv1 = bias ? bias[ncol1] : 0.0f;
  #pragma unroll
  for (int r = 0; r < 8; ++r) {
    int mrow = m0 + wm * 16 + r + 8 * (lane >> 4);
    if (mrow < M) {
      float v0 = acc0[r] + bv0, v1 = acc1[r] + bv1;
      if (act == 1) { v0 = fmaxf(v0, 0.0f); v1 = fmaxf(v1, 0.0f); }
      C[(size_t)mrow * N + ncol0] = v0;
      C[(size_t)mrow * N + ncol1] = v1;
    }
  }
}

// ---------------- elementwise / reduction kernels ----------------
__global__ void k_zero(float* p, size_t n){
  size_t i = (size_t)blockIdx.x*256 + threadIdx.x; if (i < n) p[i] = 0.f;
}

__global__ void k_split_obs(const float* __restrict__ agvs, float* __restrict__ cmain,
                            float* __restrict__ cagv){
  size_t i = (size_t)blockIdx.x*256 + threadIdx.x;
  size_t tot = (size_t)B_*NAGV_*NF_; if (i >= tot) return;
  int f = (int)(i % NF_); int a = (int)((i / NF_) % NAGV_); int b = (int)(i / ((size_t)NF_*NAGV_));
  float v = agvs[i];
  if (a == 0) cmain[(size_t)b*NF_ + f] = v;
  else        cagv[((size_t)b*(NAGV_-1) + (a-1))*NF_ + f] = v;
}

__global__ void k_pack_objects(const float* __restrict__ em, const float* __restrict__ ea,
                               float* __restrict__ obj){
  size_t i = (size_t)blockIdx.x*256 + threadIdx.x;
  size_t tot = (size_t)B_*NAGV_*EMB_; if (i >= tot) return;
  int c = (int)(i & 127); int a = (int)((i >> 7) % NAGV_); int b = (int)(i / ((size_t)NAGV_*EMB_));
  obj[i] = (a == 0) ? em[(size_t)b*EMB_ + c]
                    : ea[((size_t)b*(NAGV_-1) + (a-1))*EMB_ + c];
}

__global__ void k_posenc_nodes(const float* __restrict__ nodes, float* __restrict__ pe){
  int n = blockIdx.x*256 + threadIdx.x; if (n >= NNODE_) return;
  float x0 = nodes[n*2], x1 = nodes[n*2+1];
  const float TP = 6.283185307179586f;
  pe[n*4+0]=x0; pe[n*4+1]=x1; pe[n*4+2]=__sinf(TP*x0); pe[n*4+3]=__cosf(TP*x1);
}

__global__ void k_posenc_edges(const float* __restrict__ nodes, const int* __restrict__ src,
                               const int* __restrict__ dst, float* __restrict__ pe){
  int e = blockIdx.x*256 + threadIdx.x; if (e >= NEDGE_) return;
  int s = src[e], d = dst[e];
  float s0=nodes[s*2], s1=nodes[s*2+1], d0=nodes[d*2], d1=nodes[d*2+1];
  const float TP = 6.283185307179586f;
  float* o = pe + (size_t)e*8;
  o[0]=s0;o[1]=s1;o[2]=d0;o[3]=d1;
  o[4]=__sinf(TP*s0);o[5]=__cosf(TP*s1);o[6]=__sinf(TP*d0);o[7]=__cosf(TP*d1);
}

__global__ void k_tile128(const float* __restrict__ base, float* __restrict__ X,
                          int L, size_t tot){
  size_t i = (size_t)blockIdx.x*256 + threadIdx.x; if (i >= tot) return;
  X[i] = base[i % ((size_t)L*EMB_)];
}

__global__ void k_add_pos(const float* __restrict__ X, const float* __restrict__ pos,
                          float* __restrict__ T, int L, size_t tot){
  size_t i = (size_t)blockIdx.x*256 + threadIdx.x; if (i >= tot) return;
  T[i] = X[i] + pos[i % ((size_t)L*EMB_)];
}

// online-softmax cross-attention over 30-slot memory; one thread = (b,l,h)
__global__ __launch_bounds__(128) void k_attn(
    const float* __restrict__ Q, const float* __restrict__ K,
    const float* __restrict__ V, float* __restrict__ O, int L)
{
  long long t = (long long)blockIdx.x*128 + threadIdx.x;
  long long total = (long long)B_*L*HEADS_; if (t >= total) return;
  int h = (int)(t & 3);
  long long bl = t >> 2;              // b*L + l
  int b = (int)(bl / L);
  float qr[HD_];
  const float* q = Q + (size_t)bl*EMB_ + h*HD_;
  #pragma unroll
  for (int i = 0; i < HD_; ++i) qr[i] = q[i];
  const float scale = 0.17677669529663687f;  // 1/sqrt(32)
  float mx = -3.4e38f, sum = 0.f, o[HD_];
  #pragma unroll
  for (int i = 0; i < HD_; ++i) o[i] = 0.f;
  for (int m = 0; m < MEMN_; ++m) {
    const float* kr = K + ((size_t)b*MEMN_ + m)*EMB_ + h*HD_;
    const float* vr = V + ((size_t)b*MEMN_ + m)*EMB_ + h*HD_;
    float d = 0.f;
    #pragma unroll
    for (int i = 0; i < HD_; ++i) d += qr[i]*kr[i];
    d *= scale;
    float nm  = fmaxf(mx, d);
    float cor = __expf(mx - nm);
    float e   = __expf(d - nm);
    sum = sum*cor + e;
    #pragma unroll
    for (int i = 0; i < HD_; ++i) o[i] = o[i]*cor + e*vr[i];
    mx = nm;
  }
  float inv = 1.f/sum;
  float* op = O + (size_t)bl*EMB_ + h*HD_;
  #pragma unroll
  for (int i = 0; i < HD_; ++i) op[i] = o[i]*inv;
}

// Out = LayerNorm(Xa + Xb)*g + be ; one wave (32 lanes) per 128-wide row
__global__ __launch_bounds__(256) void k_resid_ln(
    const float* __restrict__ Xa, const float* __restrict__ Xb,
    const float* __restrict__ g, const float* __restrict__ be,
    float* __restrict__ Out, size_t R)
{
  size_t row = (size_t)blockIdx.x*8 + (threadIdx.x >> 5);
  if (row >= R) return;
  int lane = threadIdx.x & 31;
  const float* pa = Xa + row*EMB_;
  const float* pb = Xb + row*EMB_;
  float v[4]; float s = 0.f;
  #pragma unroll
  for (int i = 0; i < 4; ++i) { v[i] = pa[lane*4+i] + pb[lane*4+i]; s += v[i]; }
  #pragma unroll
  for (int o = 16; o; o >>= 1) s += __shfl_xor(s, o, 32);
  float mean = s * (1.f/EMB_);
  float q = 0.f;
  #pragma unroll
  for (int i = 0; i < 4; ++i) { float d = v[i]-mean; q += d*d; }
  #pragma unroll
  for (int o = 16; o; o >>= 1) q += __shfl_xor(q, o, 32);
  float inv = rsqrtf(q * (1.f/EMB_) + EPS_);
  #pragma unroll
  for (int i = 0; i < 4; ++i)
    Out[row*EMB_ + lane*4+i] = (v[i]-mean)*inv*g[lane*4+i] + be[lane*4+i];
}

// out[r] = dot(A[r,0:128], vec); one wave per row
__global__ __launch_bounds__(256) void k_rowdot(
    const float* __restrict__ A, const float* __restrict__ vec,
    float* __restrict__ out, size_t R)
{
  size_t row = (size_t)blockIdx.x*8 + (threadIdx.x >> 5);
  if (row >= R) return;
  int lane = threadIdx.x & 31;
  const float* a = A + row*EMB_;
  float s = 0.f;
  #pragma unroll
  for (int i = 0; i < 4; ++i) s += a[lane*4+i]*vec[lane*4+i];
  #pragma unroll
  for (int o = 16; o; o >>= 1) s += __shfl_xor(s, o, 32);
  if (!lane) out[row] = s;
}

// ---------------- GAT scatter kernels ----------------
__global__ void k_deg(const int* __restrict__ dst, float* __restrict__ deg){
  int e = blockIdx.x*256 + threadIdx.x; if (e >= NEDGE_) return;
  atomicAdd(&deg[dst[e]], 1.0f);
}

__global__ void k_eself_scatter(const float* __restrict__ ebuf, const int* __restrict__ dst,
                                float* __restrict__ eself, size_t tot){
  size_t i = (size_t)blockIdx.x*256 + threadIdx.x; if (i >= tot) return;
  int c = (int)(i & 127); int e = (int)((i >> 7) % NEDGE_);
  int b = (int)(i / ((size_t)NEDGE_*EMB_));
  atomicAdd(&eself[((size_t)b*NNODE_ + dst[e])*EMB_ + c], ebuf[i]);
}

__global__ void k_divdeg(float* __restrict__ eself, const float* __restrict__ deg, size_t tot){
  size_t i = (size_t)blockIdx.x*256 + threadIdx.x; if (i >= tot) return;
  int n = (int)((i >> 7) % NNODE_);
  eself[i] /= fmaxf(deg[n], 1.0f);
}

__global__ void k_lgself(const float* as, const float* ad, const float* ae_,
                         float* lg, float* m, size_t R){
  size_t i = (size_t)blockIdx.x*256 + threadIdx.x; if (i >= R) return;
  float v = leaky_(as[i] + ad[i] + ae_[i]);
  lg[i] = v; m[i] = v;
}

__global__ void k_lgreal_max(const float* as, const float* ad, const float* ae_,
                             const int* __restrict__ src, const int* __restrict__ dst,
                             float* er, float* m, size_t R){
  size_t t = (size_t)blockIdx.x*256 + threadIdx.x; if (t >= R) return;
  int e = (int)(t % NEDGE_); int b = (int)(t / NEDGE_);
  float v = leaky_(as[(size_t)b*NNODE_ + src[e]] + ad[(size_t)b*NNODE_ + dst[e]] + ae_[t]);
  er[t] = v;
  atomicMaxF(&m[(size_t)b*NNODE_ + dst[e]], v);
}

__global__ void k_es(const float* lg, const float* m, float* es, float* denom, size_t R){
  size_t i = (size_t)blockIdx.x*256 + threadIdx.x; if (i >= R) return;
  float v = __expf(lg[i] - m[i]); es[i] = v; denom[i] = v;
}

__global__ void k_er(float* er, const float* m, float* denom,
                     const int* __restrict__ dst, size_t R){
  size_t t = (size_t)blockIdx.x*256 + threadIdx.x; if (t >= R) return;
  int e = (int)(t % NEDGE_); int b = (int)(t / NEDGE_);
  float v = __expf(er[t] - m[(size_t)b*NNODE_ + dst[e]]);
  er[t] = v;
  atomicAdd(&denom[(size_t)b*NNODE_ + dst[e]], v);
}

__global__ void k_wreal(float* er, const float* denom, const int* __restrict__ dst, size_t R){
  size_t t = (size_t)blockIdx.x*256 + threadIdx.x; if (t >= R) return;
  int e = (int)(t % NEDGE_); int b = (int)(t / NEDGE_);
  er[t] /= denom[(size_t)b*NNODE_ + dst[e]];
}

__global__ void k_outself(const float* __restrict__ h, const float* __restrict__ es,
                          const float* __restrict__ denom, const float* __restrict__ bias,
                          float* __restrict__ out, size_t tot){
  size_t i = (size_t)blockIdx.x*256 + threadIdx.x; if (i >= tot) return;
  int c = (int)(i & 127);
  size_t rn = i >> 7;  // b*NNODE_ + n
  out[i] = (es[rn]/denom[rn]) * h[i] + bias[c];
}

__global__ void k_outscatter(const float* __restrict__ h, const float* __restrict__ w,
                             const int* __restrict__ src, const int* __restrict__ dst,
                             float* __restrict__ out, size_t tot){
  size_t i = (size_t)blockIdx.x*256 + threadIdx.x; if (i >= tot) return;
  int c = (int)(i & 127); int e = (int)((i >> 7) % NEDGE_);
  int b = (int)(i / ((size_t)NEDGE_*EMB_));
  atomicAdd(&out[((size_t)b*NNODE_ + dst[e])*EMB_ + c],
            w[(size_t)b*NEDGE_ + e] * h[((size_t)b*NNODE_ + src[e])*EMB_ + c]);
}

__global__ __launch_bounds__(256) void k_bn_stats(const float* __restrict__ X,
                                                  float* mean, float* var, int R){
  int c = blockIdx.x;                // channel 0..127
  __shared__ float ss[256], sq[256];
  float s = 0.f, q = 0.f;
  for (int r = threadIdx.x; r < R; r += 256){
    float v = X[(size_t)r*EMB_ + c]; s += v; q += v*v;
  }
  ss[threadIdx.x] = s; sq[threadIdx.x] = q; __syncthreads();
  for (int o = 128; o; o >>= 1){
    if ((int)threadIdx.x < o){ ss[threadIdx.x]+=ss[threadIdx.x+o]; sq[threadIdx.x]+=sq[threadIdx.x+o]; }
    __syncthreads();
  }
  if (!threadIdx.x){ float m = ss[0]/R; mean[c] = m; var[c] = sq[0]/R - m*m; }
}

__global__ void k_bn_apply(const float* __restrict__ X, const float* mean, const float* var,
                           const float* g, const float* b, float* __restrict__ O, size_t tot){
  size_t i = (size_t)blockIdx.x*256 + threadIdx.x; if (i >= tot) return;
  int c = (int)(i & 127);
  O[i] = (X[i]-mean[c]) * rsqrtf(var[c]+EPS_) * g[c] + b[c];
}

__global__ void k_inreach(const float* __restrict__ agvs, const float* __restrict__ nodes,
                          int* __restrict__ reach){
  int t = blockIdx.x*256 + threadIdx.x; if (t >= B_*4) return;
  int b = t >> 2, j = t & 3;
  float c0 = agvs[(size_t)b*NAGV_*NF_ + 8 + j*2];
  float c1 = agvs[(size_t)b*NAGV_*NF_ + 8 + j*2 + 1];
  float best = 3.4e38f; int bi = 0;
  for (int n = 0; n < NNODE_; ++n){
    float dx = c0 - nodes[n*2], dy = c1 - nodes[n*2+1];
    float d = dx*dx + dy*dy;
    if (d < best){ best = d; bi = n; }
  }
  reach[t] = bi;
}

__global__ void k_gather_out(const float* __restrict__ X, const int* __restrict__ reach,
                             float* __restrict__ out){
  size_t i = (size_t)blockIdx.x*256 + threadIdx.x;
  size_t tot = (size_t)B_*4*EMB_; if (i >= tot) return;
  int c = (int)(i & 127); int j = (int)((i >> 7) & 3); int b = (int)(i >> 9);
  out[i] = X[((size_t)b*NNODE_ + reach[b*4+j])*EMB_ + c];
}

// ---------------- host-side helpers ----------------
static inline dim3 ew(size_t n){ return dim3((unsigned)((n + 255) / 256)); }

static inline void launch_gemm(const float* A, const float* W, const float* bias, float* C,
                               int M, int N, int K, int act, hipStream_t s){
  dim3 grid((N + 63)/64, (M + 63)/64);
  k_gemm<<<grid, 256, 0, s>>>(A, W, bias, C, M, N, K, act);
}

struct DecP { const float* p[17]; };
// leaf order: pos Wq bq Wk bk Wv bv Wo bo g1 b1 Wf1 bf1 Wf2 bf2 g2 b2
enum { D_POS=0,D_WQ,D_BQ,D_WK,D_BK,D_WV,D_BV,D_WO,D_BO,D_G1,D_B1,D_WF1,D_BF1,D_WF2,D_BF2,D_G2,D_B2 };

static void run_decoder(const DecP& P, const float* objects, float* X, int L,
                        float* T1, float* T2, float* H, float* kbuf, float* vbuf,
                        hipStream_t s){
  size_t rows = (size_t)B_*L, tot = rows*EMB_;
  k_add_pos<<<ew(tot),256,0,s>>>(X, P.p[D_POS], T1, L, tot);
  launch_gemm(T1, P.p[D_WQ], P.p[D_BQ], T2, (int)rows, EMB_, EMB_, 0, s);
  launch_gemm(objects, P.p[D_WK], P.p[D_BK], kbuf, B_*MEMN_, EMB_, EMB_, 0, s);
  launch_gemm(objects, P.p[D_WV], P.p[D_BV], vbuf, B_*MEMN_, EMB_, EMB_, 0, s);
  size_t nthr = (size_t)B_*L*HEADS_;
  k_attn<<<dim3((unsigned)((nthr+127)/128)),128,0,s>>>(T2, kbuf, vbuf, T1, L);
  launch_gemm(T1, P.p[D_WO], P.p[D_BO], T2, (int)rows, EMB_, EMB_, 0, s);
  k_resid_ln<<<ew(rows*32),256,0,s>>>(X, T2, P.p[D_G1], P.p[D_B1], T1, rows);
  for (size_t r0 = 0; r0 < rows; r0 += HCH_){
    int mr = (int)((rows - r0) < (size_t)HCH_ ? (rows - r0) : (size_t)HCH_);
    launch_gemm(T1 + r0*EMB_, P.p[D_WF1], P.p[D_BF1], H, mr, 4*EMB_, EMB_, 1, s);
    launch_gemm(H, P.p[D_WF2], P.p[D_BF2], T2 + r0*EMB_, mr, EMB_, 4*EMB_, 0, s);
  }
  k_resid_ln<<<ew(rows*32),256,0,s>>>(T1, T2, P.p[D_G2], P.p[D_B2], X, rows);
}

extern "C" void kernel_launch(void* const* d_in, const int* in_sizes, int n_in,
                              void* d_out, int out_size, void* d_ws, size_t ws_size,
                              hipStream_t stream)
{
  (void)in_sizes; (void)n_in; (void)out_size; (void)ws_size;
  #define PF(i) ((const float*)d_in[(i)])
  const float* agvs = PF(0);
  // params flattened in insertion order:
  // 1..4 emb_main, 5..8 emb_agv, 9..12 emb_node, 13..16 emb_edge,
  // 17..50 node_att (2x17), 51..84 edge_att (2x17), 85..120 gat (6x6), 121..132 bn (6x2)
  const float* nodes = PF(133);
  const int*   ei    = (const int*)d_in[134];
  const int*   esrc  = ei;
  const int*   edst  = ei + NEDGE_;

  // --- bump allocator over workspace ---
  char* wp = (char*)d_ws;
  auto alloc = [&](size_t nf)->float*{
    float* r = (float*)wp;
    wp += ((nf*sizeof(float) + 255) & ~(size_t)255);
    return r;
  };
  float* cmain  = alloc((size_t)B_*NF_);
  float* cagv   = alloc((size_t)B_*(NAGV_-1)*NF_);
  float* hmain  = alloc((size_t)B_*2*EMB_);
  float* hagv   = alloc((size_t)B_*(NAGV_-1)*2*EMB_);
  float* emain  = alloc((size_t)B_*EMB_);
  float* eagv   = alloc((size_t)B_*(NAGV_-1)*EMB_);
  float* objects= alloc((size_t)B_*NAGV_*EMB_);
  float* pen    = alloc((size_t)NNODE_*4);
  float* hnode  = alloc((size_t)NNODE_*2*EMB_);
  float* bnode  = alloc((size_t)NNODE_*EMB_);
  float* pee    = alloc((size_t)NEDGE_*8);
  float* hedge  = alloc((size_t)NEDGE_*2*EMB_);
  float* bedge  = alloc((size_t)NEDGE_*EMB_);
  float* Xn     = alloc((size_t)B_*NNODE_*EMB_);
  float* Xe     = alloc((size_t)B_*NEDGE_*EMB_);
  float* T1     = alloc((size_t)B_*NEDGE_*EMB_);
  float* T2     = alloc((size_t)B_*NEDGE_*EMB_);
  float* H      = alloc((size_t)HCH_*4*EMB_);
  float* kbuf   = alloc((size_t)B_*MEMN_*EMB_);
  float* vbuf   = alloc((size_t)B_*MEMN_*EMB_);
  float* deg    = alloc(NNODE_);
  float* asrcb  = alloc((size_t)B_*NNODE_);
  float* adstb  = alloc((size_t)B_*NNODE_);
  float* aesb   = alloc((size_t)B_*NNODE_);
  float* lgself = alloc((size_t)B_*NNODE_);
  float* mbuf   = alloc((size_t)B_*NNODE_);
  float* esbuf  = alloc((size_t)B_*NNODE_);
  float* denomb = alloc((size_t)B_*NNODE_);
  float* aeb    = alloc((size_t)B_*NEDGE_);
  float* erbuf  = alloc((size_t)B_*NEDGE_);
  float* bnm    = alloc(EMB_);
  float* bnv    = alloc(EMB_);
  int*   reach  = (int*)alloc((size_t)B_*4);
  // GAT-phase reuse of decoder temporaries (decoders finished by then)
  float* hbuf   = T1;
  float* eself  = T1 + (size_t)B_*NNODE_*EMB_;
  float* outbuf = eself + (size_t)B_*NNODE_*EMB_;
  float* ebuf   = T2;

  // ---------- embeddings ----------
  k_split_obs<<<ew((size_t)B_*NAGV_*NF_),256,0,stream>>>(agvs, cmain, cagv);
  launch_gemm(cmain, PF(1), PF(2), hmain, B_, 2*EMB_, NF_, 1, stream);
  launch_gemm(hmain, PF(3), PF(4), emain, B_, EMB_, 2*EMB_, 1, stream);
  launch_gemm(cagv,  PF(5), PF(6), hagv,  B_*(NAGV_-1), 2*EMB_, NF_, 1, stream);
  launch_gemm(hagv,  PF(7), PF(8), eagv,  B_*(NAGV_-1), EMB_, 2*EMB_, 1, stream);
  k_pack_objects<<<ew((size_t)B_*NAGV_*EMB_),256,0,stream>>>(emain, eagv, objects);

  k_posenc_nodes<<<ew(NNODE_),256,0,stream>>>(nodes, pen);
  launch_gemm(pen,   PF(9),  PF(10), hnode, NNODE_, 2*EMB_, 4, 1, stream);
  launch_gemm(hnode, PF(11), PF(12), bnode, NNODE_, EMB_, 2*EMB_, 1, stream);
  k_posenc_edges<<<ew(NEDGE_),256,0,stream>>>(nodes, esrc, edst, pee);
  launch_gemm(pee,   PF(13), PF(14), hedge, NEDGE_, 2*EMB_, 8, 1, stream);
  launch_gemm(hedge, PF(15), PF(16), bedge, NEDGE_, EMB_, 2*EMB_, 1, stream);
  k_tile128<<<ew((size_t)B_*NNODE_*EMB_),256,0,stream>>>(bnode, Xn, NNODE_, (size_t)B_*NNODE_*EMB_);
  k_tile128<<<ew((size_t)B_*NEDGE_*EMB_),256,0,stream>>>(bedge, Xe, NEDGE_, (size_t)B_*NEDGE_*EMB_);

  // ---------- decoder blocks ----------
  for (int i = 0; i < 2; ++i){
    DecP P; for (int j = 0; j < 17; ++j) P.p[j] = PF(17 + 17*i + j);
    run_decoder(P, objects, Xn, NNODE_, T1, T2, H, kbuf, vbuf, stream);
  }
  for (int i = 0; i < 2; ++i){
    DecP P; for (int j = 0; j < 17; ++j) P.p[j] = PF(51 + 17*i + j);
    run_decoder(P, objects, Xe, NEDGE_, T1, T2, H, kbuf, vbuf, stream);
  }

  // ---------- GAT layers ----------
  k_zero<<<ew(NNODE_),256,0,stream>>>(deg, NNODE_);
  k_deg<<<ew(NEDGE_),256,0,stream>>>(edst, deg);
  const size_t RN = (size_t)B_*NNODE_, RE = (size_t)B_*NEDGE_;
  for (int i = 0; i < 6; ++i){
    const float* W   = PF(85 + 6*i + 0);
    const float* We  = PF(85 + 6*i + 1);
    const float* asv = PF(85 + 6*i + 2);
    const float* adv = PF(85 + 6*i + 3);
    const float* aev = PF(85 + 6*i + 4);
    const float* gbia= PF(85 + 6*i + 5);
    const float* bng = PF(121 + 2*i + 0);
    const float* bnb = PF(121 + 2*i + 1);
    launch_gemm(Xn, W,  nullptr, hbuf, (int)RN, EMB_, EMB_, 0, stream);
    launch_gemm(Xe, We, nullptr, ebuf, (int)RE, EMB_, EMB_, 0, stream);
    k_zero<<<ew(RN*EMB_),256,0,stream>>>(eself, RN*EMB_);
    k_eself_scatter<<<ew(RE*EMB_),256,0,stream>>>(ebuf, edst, eself, RE*EMB_);
    k_divdeg<<<ew(RN*EMB_),256,0,stream>>>(eself, deg, RN*EMB_);
    k_rowdot<<<ew(RN*32),256,0,stream>>>(hbuf,  asv, asrcb, RN);
    k_rowdot<<<ew(RN*32),256,0,stream>>>(hbuf,  adv, adstb, RN);
    k_rowdot<<<ew(RE*32),256,0,stream>>>(ebuf,  aev, aeb,   RE);
    k_rowdot<<<ew(RN*32),256,0,stream>>>(eself, aev, aesb,  RN);
    k_lgself<<<ew(RN),256,0,stream>>>(asrcb, adstb, aesb, lgself, mbuf, RN);
    k_lgreal_max<<<ew(RE),256,0,stream>>>(asrcb, adstb, aeb, esrc, edst, erbuf, mbuf, RE);
    k_es<<<ew(RN),256,0,stream>>>(lgself, mbuf, esbuf, denomb, RN);
    k_er<<<ew(RE),256,0,stream>>>(erbuf, mbuf, denomb, edst, RE);
    k_wreal<<<ew(RE),256,0,stream>>>(erbuf, denomb, edst, RE);
    k_outself<<<ew(RN*EMB_),256,0,stream>>>(hbuf, esbuf, denomb, gbia, outbuf, RN*EMB_);
    k_outscatter<<<ew(RE*EMB_),256,0,stream>>>(hbuf, erbuf, esrc, edst, outbuf, RE*EMB_);
    k_bn_stats<<<dim3(EMB_),256,0,stream>>>(outbuf, bnm, bnv, (int)RN);
    k_bn_apply<<<ew(RN*EMB_),256,0,stream>>>(outbuf, bnm, bnv, bng, bnb, Xn, RN*EMB_);
  }

  // ---------- gather output ----------
  k_inreach<<<ew((size_t)B_*4),256,0,stream>>>(agvs, nodes, reach);
  k_gather_out<<<ew((size_t)B_*4*EMB_),256,0,stream>>>(Xn, reach, (float*)d_out);
  #undef PF
}
